// FeatureMemory_11244224381195
// MI455X (gfx1250) — matrix operand order repaired and checked by reading, not compiled, not run on hardware
//
#include <hip/hip_runtime.h>
#include <hip/hip_bf16.h>

// Problem constants
#define CDIM 512
#define MDIM 1024
#define NDIM 4096       // H*W
#define BATCH 16
#define NT 64           // columns per workgroup
#define MC 128          // m-chunk per online-softmax iteration
#define XS 520          // x_lds row stride (bf16 elems), 16B aligned, bank-spread
#define SC 136          // s_chunk row stride (f32 elems), 16B aligned
#define PC 136          // p_chunk row stride (bf16 elems), 16B aligned

typedef __bf16 bf16_t;
typedef bf16_t v8bf  __attribute__((ext_vector_type(8)));
typedef bf16_t v16bf __attribute__((ext_vector_type(16)));
typedef float  v8f   __attribute__((ext_vector_type(8)));

static __device__ __forceinline__ unsigned short f2bf(float f) {
    unsigned int u = __builtin_bit_cast(unsigned int, f);
    u += 0x7FFFu + ((u >> 16) & 1u);           // round-to-nearest-even
    return (unsigned short)(u >> 16);
}

static __device__ __forceinline__ v16bf mk_op(const unsigned short* p0,
                                              const unsigned short* p1) {
    v8bf a = *(const v8bf*)p0;
    v8bf b = *(const v8bf*)p1;
    return __builtin_shufflevector(a, b, 0,1,2,3,4,5,6,7,8,9,10,11,12,13,14,15);
}

#define WMMA_BF16(A, B, Cacc) \
    __builtin_amdgcn_wmma_f32_16x16x32_bf16(false, (A), false, (B), (short)0, (Cacc), false, false)

// ---------------- prep: bf16 mem [C][M] and memT [M][C] in workspace ---------
__global__ void fm_prep(const float* __restrict__ mem,
                        unsigned short* __restrict__ memT,
                        unsigned short* __restrict__ memB) {
    int idx = blockIdx.x * blockDim.x + threadIdx.x;   // over C*M
    int c = idx >> 10;
    int m = idx & (MDIM - 1);
    unsigned short v = f2bf(mem[idx]);
    memB[idx] = v;                 // [C][M] row-major (GEMM2 A)
    memT[m * CDIM + c] = v;        // [M][C] row-major (GEMM1 A)
}

// ------------- fused flash-style: GEMM1 -> online softmax -> GEMM2 ----------
__global__ void __launch_bounds__(256)
fm_fused(const float* __restrict__ x,
         const unsigned short* __restrict__ memT,   // [M][C] bf16
         const unsigned short* __restrict__ memB,   // [C][M] bf16
         float* __restrict__ out) {
    __shared__ unsigned short x_lds[NT * XS];       // x tile, [n][c] bf16      (~66.5 KB)
    __shared__ float          s_chunk[NT * SC];     // score chunk [n][m] f32   (~34.8 KB)
    __shared__ unsigned short p_chunk[NT * PC];     // exp chunk [n][m] bf16    (~17.4 KB)
    __shared__ float          red[NT * 4];          // per-column partials
    __shared__ float          rmax[NT];             // running max
    __shared__ float          rsum[NT];             // running sum
    __shared__ float          alpha[NT];            // rescale factor / final 1/sum

    const int t    = threadIdx.x;
    const int lane = t & 31;
    const int wave = t >> 5;                        // 0..7
    const int half = lane >> 4;
    const int l15  = lane & 15;

    const int bidx = blockIdx.x >> 6;               // N/NT = 64 tiles per batch
    const int n0   = (blockIdx.x & 63) * NT;

    // ---- stage x tile -> LDS, transposed, bf16 -----------------------------
    const float* xb = x + (size_t)bidx * CDIM * NDIM;
    {
        const int j  = t & 63;                      // column within tile
        const int cb = t >> 6;                      // 0..3
        #pragma unroll 4
        for (int i = 0; i < 128; ++i) {
            int c = cb + (i << 2);
            x_lds[j * XS + c] = f2bf(xb[(size_t)c * NDIM + n0 + j]);
        }
    }
    if (t < NT) { rmax[t] = -3.0e38f; rsum[t] = 0.0f; }

    // persistent output accumulator: wave owns c rows [64*wave, 64*wave+64)
    v8f O[4][4];                                    // [c-rowtile][n-tile]
    #pragma unroll
    for (int ct = 0; ct < 4; ++ct)
        #pragma unroll
        for (int jn = 0; jn < 4; ++jn) O[ct][jn] = v8f{};

    __syncthreads();

    const int scol   = t & 63;                      // softmax column
    const int sslice = t >> 6;                      // 0..3, 32 m each

    for (int it = 0; it < 8; ++it) {
        const int m0 = it << 7;                     // chunk base in M

        // ---- GEMM1 chunk: S[m,n], wave computes rows [m0+16w, +16) ---------
        {
            v8f S[4] = {v8f{}, v8f{}, v8f{}, v8f{}};
            const unsigned short* Arow =
                memT + (size_t)(m0 + (wave << 4) + l15) * CDIM;
            #pragma unroll 4
            for (int kk = 0; kk < 16; ++kk) {
                const int c0 = kk << 5;
                v16bf A = mk_op(Arow + c0 + 8 * half, Arow + c0 + 16 + 8 * half);
                #pragma unroll
                for (int jn = 0; jn < 4; ++jn) {
                    const unsigned short* bp =
                        &x_lds[((jn << 4) + l15) * XS + c0 + 8 * half];
                    v16bf B = mk_op(bp, bp + 16);
                    S[jn] = WMMA_BF16(A, B, S[jn]);
                }
            }
            #pragma unroll
            for (int jn = 0; jn < 4; ++jn)
                #pragma unroll
                for (int v = 0; v < 8; ++v)
                    s_chunk[((jn << 4) + l15) * SC + (wave << 4) + v + 8 * half] = S[jn][v];
        }
        __syncthreads();

        // ---- online softmax over this chunk --------------------------------
        {
            const float* srow = &s_chunk[scol * SC + sslice * 32];
            float mx = -3.0e38f;
            #pragma unroll 8
            for (int i = 0; i < 8; ++i) {
                float4 v = ((const float4*)srow)[i];
                mx = fmaxf(mx, fmaxf(fmaxf(v.x, v.y), fmaxf(v.z, v.w)));
            }
            red[scol * 4 + sslice] = mx;
        }
        __syncthreads();
        if (t < NT) {
            float mx = fmaxf(fmaxf(red[t * 4], red[t * 4 + 1]),
                             fmaxf(red[t * 4 + 2], red[t * 4 + 3]));
            float nm = fmaxf(rmax[t], mx);
            alpha[t] = __expf(rmax[t] - nm);        // rescale for old state
            rmax[t]  = nm;
        }
        __syncthreads();
        {
            const float cmax = rmax[scol];
            const float* srow = &s_chunk[scol * SC + sslice * 32];
            unsigned int* prow = (unsigned int*)&p_chunk[scol * PC + sslice * 32];
            float sum = 0.0f;
            #pragma unroll 8
            for (int i = 0; i < 8; ++i) {
                float4 v = ((const float4*)srow)[i];
                float e0 = __expf(v.x - cmax), e1 = __expf(v.y - cmax);
                float e2 = __expf(v.z - cmax), e3 = __expf(v.w - cmax);
                sum += (e0 + e1) + (e2 + e3);
                prow[2 * i]     = (unsigned int)f2bf(e0) | ((unsigned int)f2bf(e1) << 16);
                prow[2 * i + 1] = (unsigned int)f2bf(e2) | ((unsigned int)f2bf(e3) << 16);
            }
            red[scol * 4 + sslice] = sum;
        }
        __syncthreads();
        if (t < NT) {
            rsum[t] = rsum[t] * alpha[t] +
                      ((red[t * 4] + red[t * 4 + 1]) + (red[t * 4 + 2] + red[t * 4 + 3]));
        }
        __syncthreads();

        // ---- rescale O and accumulate GEMM2 over this chunk ----------------
        #pragma unroll
        for (int jn = 0; jn < 4; ++jn) {
            const float a = alpha[(jn << 4) + l15];
            #pragma unroll
            for (int ct = 0; ct < 4; ++ct)
                #pragma unroll
                for (int v = 0; v < 8; ++v) O[ct][jn][v] *= a;
        }
        const int cw0 = wave << 6;                  // wave's first c row
        #pragma unroll
        for (int kk = 0; kk < 4; ++kk) {
            const int k0 = kk << 5;
            // preload ALL operand tiles for this k-step: 8 global_load_b128
            // (A) + 8 ds_load_b128 (B) outstanding -> graduated waits, no
            // per-rowtile load-use stall.
            v16bf At[4];
            v16bf Bt[4];
            #pragma unroll
            for (int ct = 0; ct < 4; ++ct) {
                const unsigned short* Arow =
                    memB + (size_t)(cw0 + (ct << 4) + l15) * MDIM + m0 + k0;
                At[ct] = mk_op(Arow + 8 * half, Arow + 16 + 8 * half);
            }
            #pragma unroll
            for (int jn = 0; jn < 4; ++jn) {
                const unsigned short* bp =
                    &p_chunk[((jn << 4) + l15) * PC + k0 + 8 * half];
                Bt[jn] = mk_op(bp, bp + 16);
            }
            #pragma unroll
            for (int ct = 0; ct < 4; ++ct)
                #pragma unroll
                for (int jn = 0; jn < 4; ++jn)
                    O[ct][jn] = WMMA_BF16(At[ct], Bt[jn], O[ct][jn]);
        }
        __syncthreads();                            // p/s chunks reusable next iter
    }

    // ---- finalize: scale by 1/sum and store --------------------------------
    if (t < NT) alpha[t] = 1.0f / rsum[t];
    __syncthreads();

    const size_t obase = (size_t)bidx * CDIM * NDIM + n0;
    #pragma unroll
    for (int jn = 0; jn < 4; ++jn) {
        const float sc = alpha[(jn << 4) + l15];
        #pragma unroll
        for (int ct = 0; ct < 4; ++ct) {
            #pragma unroll
            for (int v = 0; v < 8; ++v) {
                const int c = (wave << 6) + (ct << 4) + v + 8 * half;
                out[obase + (size_t)c * NDIM + (jn << 4) + l15] = O[ct][jn][v] * sc;
            }
        }
    }
}

extern "C" void kernel_launch(void* const* d_in, const int* in_sizes, int n_in,
                              void* d_out, int out_size, void* d_ws, size_t ws_size,
                              hipStream_t stream) {
    const float* x   = (const float*)d_in[0];
    const float* mem = (const float*)d_in[1];
    float* out = (float*)d_out;

    unsigned short* memT = (unsigned short*)d_ws;            // [M][C] bf16, 1 MiB
    unsigned short* memB = memT + (size_t)CDIM * MDIM;       // [C][M] bf16, 1 MiB

    fm_prep<<<(CDIM * MDIM) / 256, 256, 0, stream>>>(mem, memT, memB);
    fm_fused<<<BATCH * (NDIM / NT), 256, 0, stream>>>(x, memT, memB, out);
}
